// GPTAlpha_Tmix_20830591385838
// MI455X (gfx1250) — matrix operand order repaired and checked by reading, not compiled, not run on hardware
//
#include <hip/hip_runtime.h>
#include <hip/hip_bf16.h>
#include <math.h>

// ---------------- CDNA5 types ----------------
typedef __bf16 bf16;
typedef __attribute__((ext_vector_type(8)))  bf16  v8bf;
typedef __attribute__((ext_vector_type(16))) bf16  v16bf;
typedef __attribute__((ext_vector_type(8)))  float v8f;

#define B_  2
#define T_  1024
#define C_  2048
#define H_  32
#define D_  64
#define M_  (B_*T_)

static __device__ __forceinline__ v16bf cat16(v8bf lo, v8bf hi) {
  return __builtin_shufflevector(lo, hi, 0,1,2,3,4,5,6,7,8,9,10,11,12,13,14,15);
}

// ---------------- Stage A: token shift ----------------
__global__ void prep_shift(const float* __restrict__ x, const float* __restrict__ shift,
                           const float* __restrict__ tmx,
                           float* __restrict__ dxprev, float* __restrict__ xxx) {
  size_t i = (size_t)blockIdx.x * blockDim.x + threadIdx.x;
  if (i >= (size_t)M_ * C_) return;
  int c = (int)(i & (C_ - 1));
  size_t tc = i % ((size_t)T_ * C_);
  size_t b  = i / ((size_t)T_ * C_);
  float xp = (tc < C_) ? shift[b * C_ + c] : x[i - C_];
  float xi = x[i];
  float dx = xp - xi;
  dxprev[i] = dx;
  xxx[i] = xi + dx * tmx[c];
}

// ---------------- Weight transpose f32(K,N) -> bf16(N,K) ----------------
__global__ __launch_bounds__(256)
void transpose_to_bf16(const float* __restrict__ in, bf16* __restrict__ out, int K, int N) {
  __shared__ float tile[32][33];
  int k0 = blockIdx.y * 32, n0 = blockIdx.x * 32;
  int tx = threadIdx.x, ty = threadIdx.y;          // block (32,8)
#pragma unroll
  for (int i = 0; i < 32; i += 8)
    tile[ty + i][tx] = in[(size_t)(k0 + ty + i) * N + n0 + tx];
  __syncthreads();
#pragma unroll
  for (int i = 0; i < 32; i += 8)
    out[(size_t)(n0 + ty + i) * K + k0 + tx] = (bf16)tile[tx][ty + i];
}

// ---------------- LoRA stage 1: h = tanh(xxx @ w1) ----------------
__global__ __launch_bounds__(128)
void lora1(const float* __restrict__ xxx, const float* __restrict__ w1,
           float* __restrict__ hbuf) {
  const int row = blockIdx.x;
  __shared__ float xs[C_];
  for (int c = threadIdx.x; c < C_; c += 128) xs[c] = xxx[(size_t)row * C_ + c];
  __syncthreads();
  if (threadIdx.x < 96) {
    float acc = 0.f;
    for (int c = 0; c < C_; ++c) acc += xs[c] * w1[(size_t)c * 96 + threadIdx.x];
    hbuf[row * 96 + threadIdx.x] = tanhf(acc);
  }
}

// ---------------- LoRA stage 2 + mix: xq/xk/xv (bf16 out) ----------------
__global__ __launch_bounds__(256)
void mix_qkv(const float* __restrict__ x, const float* __restrict__ dxprev,
             const float* __restrict__ hbuf, const float* __restrict__ w2,
             const float* __restrict__ tmr, const float* __restrict__ tmk,
             const float* __restrict__ tmv,
             bf16* __restrict__ xq, bf16* __restrict__ xk, bf16* __restrict__ xv) {
  const int row = blockIdx.x;
  __shared__ float hs[96];
  if (threadIdx.x < 96) hs[threadIdx.x] = hbuf[row * 96 + threadIdx.x];
  __syncthreads();
  for (int c = threadIdx.x; c < C_; c += 256) {
    float mr = 0.f, mk = 0.f, mv = 0.f;
#pragma unroll 8
    for (int i = 0; i < 32; ++i) {
      mr += hs[i]      * w2[(size_t)(i)      * C_ + c];
      mk += hs[32 + i] * w2[(size_t)(32 + i) * C_ + c];
      mv += hs[64 + i] * w2[(size_t)(64 + i) * C_ + c];
    }
    size_t idx = (size_t)row * C_ + c;
    float xx = x[idx], dx = dxprev[idx];
    xq[idx] = (bf16)(xx + dx * (tmr[c] + mr));
    xk[idx] = (bf16)(xx + dx * (tmk[c] + mk));
    xv[idx] = (bf16)(xx + dx * (tmv[c] + mv));
  }
}

// ---------------- bf16 WMMA GEMM: C(MxN,f32) = A(MxK) * Bt(NxK)^T ----------------
// One wave per 64x64 tile; 16 v_wmma_f32_16x16x32_bf16 per K-step.
__global__ __launch_bounds__(32)
void gemm_bf16(const bf16* __restrict__ A, const bf16* __restrict__ Bt,
               float* __restrict__ Cout, int Mdim, int Ndim, int Kdim) {
  const int lane = threadIdx.x;
  const int col  = lane & 15;
  const int hi   = lane >> 4;
  const int m0 = blockIdx.y * 64;
  const int n0 = blockIdx.x * 64;
  const int ka = hi ? 8 : 0;     // A-frag K offset per ISA 16-bit A layout
  const int kb = hi ? 8 : 0;     // B-frag K offset, same K order as A
  v8f acc[4][4] = {};
  for (int k0 = 0; k0 < Kdim; k0 += 32) {
    v16bf af[4], bfr[4];
#pragma unroll
    for (int mt = 0; mt < 4; ++mt) {
      const bf16* p = A + (size_t)(m0 + mt * 16 + col) * Kdim + k0 + ka;
      af[mt] = cat16(*(const v8bf*)p, *(const v8bf*)(p + 16));
    }
#pragma unroll
    for (int nt = 0; nt < 4; ++nt) {
      const bf16* p = Bt + (size_t)(n0 + nt * 16 + col) * Kdim + k0 + kb;
      bfr[nt] = cat16(*(const v8bf*)p, *(const v8bf*)(p + 16));
    }
#pragma unroll
    for (int mt = 0; mt < 4; ++mt)
#pragma unroll
      for (int nt = 0; nt < 4; ++nt)
        acc[mt][nt] = __builtin_amdgcn_wmma_f32_16x16x32_bf16(
            false, af[mt], false, bfr[nt], (short)0, acc[mt][nt], false, false);
  }
#pragma unroll
  for (int mt = 0; mt < 4; ++mt)
#pragma unroll
    for (int nt = 0; nt < 4; ++nt)
#pragma unroll
      for (int j = 0; j < 8; ++j)
        Cout[(size_t)(m0 + mt * 16 + j + 8 * hi) * Ndim + n0 + nt * 16 + col] =
            acc[mt][nt][j];
}

// ---------------- LN + RoPE -> (b,h,t,d) bf16 ----------------
__global__ __launch_bounds__(256)
void ln_rope_qk(const float* __restrict__ pre, const float* __restrict__ g,
                const float* __restrict__ bb, const float* __restrict__ cosb,
                const float* __restrict__ sinb, bf16* __restrict__ dst) {
  const int bid = blockIdx.x;
  const int t = bid & (T_ - 1);
  const int b = bid >> 10;
  const float* row = pre + (size_t)bid * C_;
  __shared__ float r1[256], r2[256];
  float s = 0.f, s2 = 0.f;
  for (int c = threadIdx.x; c < C_; c += 256) { float v = row[c]; s += v; s2 += v * v; }
  r1[threadIdx.x] = s; r2[threadIdx.x] = s2;
  __syncthreads();
  for (int st = 128; st > 0; st >>= 1) {
    if (threadIdx.x < st) { r1[threadIdx.x] += r1[threadIdx.x + st];
                            r2[threadIdx.x] += r2[threadIdx.x + st]; }
    __syncthreads();
  }
  const float mu = r1[0] / C_;
  const float rinv = rsqrtf(r2[0] / C_ - mu * mu + 1e-5f);
  for (int c = threadIdx.x; c < C_; c += 256) {
    const int h = c >> 6, d = c & 63;
    float v = (row[c] - mu) * rinv * g[c] + bb[c];
    float o;
    if (d < 32) {
      float v2 = (row[c + 32] - mu) * rinv * g[c + 32] + bb[c + 32];
      float cc = cosb[t * 32 + d], ss = sinb[t * 32 + d];
      o = v * cc - v2 * ss;
    } else {
      float v1 = (row[c - 32] - mu) * rinv * g[c - 32] + bb[c - 32];
      float cc = cosb[t * 32 + d - 32], ss = sinb[t * 32 + d - 32];
      o = v1 * ss + v * cc;
    }
    dst[((size_t)(b * H_ + h) * T_ + t) * D_ + d] = (bf16)o;
  }
}

// ---------------- LN(v) -> transposed (b,h,d,t) bf16 ----------------
__global__ __launch_bounds__(256)
void ln_v(const float* __restrict__ pre, const float* __restrict__ g,
          const float* __restrict__ bb, bf16* __restrict__ vt) {
  const int bid = blockIdx.x;
  const int t = bid & (T_ - 1);
  const int b = bid >> 10;
  const float* row = pre + (size_t)bid * C_;
  __shared__ float r1[256], r2[256];
  float s = 0.f, s2 = 0.f;
  for (int c = threadIdx.x; c < C_; c += 256) { float v = row[c]; s += v; s2 += v * v; }
  r1[threadIdx.x] = s; r2[threadIdx.x] = s2;
  __syncthreads();
  for (int st = 128; st > 0; st >>= 1) {
    if (threadIdx.x < st) { r1[threadIdx.x] += r1[threadIdx.x + st];
                            r2[threadIdx.x] += r2[threadIdx.x + st]; }
    __syncthreads();
  }
  const float mu = r1[0] / C_;
  const float rinv = rsqrtf(r2[0] / C_ - mu * mu + 1e-5f);
  for (int c = threadIdx.x; c < C_; c += 256) {
    const int h = c >> 6, d = c & 63;
    float v = (row[c] - mu) * rinv * g[c] + bb[c];
    vt[((size_t)(b * H_ + h) * D_ + d) * T_ + t] = (bf16)v;
  }
}

// ---------------- Flash attention, one wave per 16 query rows ----------------
__global__ __launch_bounds__(32)
void attn_flash(const bf16* __restrict__ qbh, const bf16* __restrict__ kbh,
                const bf16* __restrict__ vt, float* __restrict__ obh) {
  const int lane = threadIdx.x;
  const int col  = lane & 15;
  const int hi   = lane >> 4;
  const int ka = hi ? 8 : 0;
  const int kb = hi ? 8 : 0;
  const int q0 = blockIdx.x * 16;
  const int bh = blockIdx.z * H_ + blockIdx.y;

  // Q A-fragments for d-chunks {0..31, 32..63}
  const bf16* qrow = qbh + ((size_t)bh * T_ + q0 + col) * D_;
  v16bf qa[2];
#pragma unroll
  for (int dc = 0; dc < 2; ++dc) {
    const bf16* p = qrow + dc * 32 + ka;
    qa[dc] = cat16(*(const v8bf*)p, *(const v8bf*)(p + 16));
  }

  __shared__ bf16 pLds[16 * 32];
  float mrun[8], lrun[8];
  v8f oacc[4] = {};
#pragma unroll
  for (int j = 0; j < 8; ++j) { mrun[j] = -1e30f; lrun[j] = 0.f; }
  const float scale = 0.125f;

  for (int s0 = 0; s0 <= q0 + 15; s0 += 32) {
    // S = Q K^T for two 16-key subtiles (K-dim 64 -> 2 wmma each)
    v8f sv[2] = {};
#pragma unroll
    for (int sub = 0; sub < 2; ++sub) {
      const bf16* krow = kbh + ((size_t)bh * T_ + s0 + sub * 16 + col) * D_;
#pragma unroll
      for (int dc = 0; dc < 2; ++dc) {
        const bf16* p = krow + dc * 32 + kb;
        v16bf kf = cat16(*(const v8bf*)p, *(const v8bf*)(p + 16));
        sv[sub] = __builtin_amdgcn_wmma_f32_16x16x32_bf16(
            false, qa[dc], false, kf, (short)0, sv[sub], false, false);
      }
    }
    // online softmax over 32 keys (row stats via 16-lane butterflies)
#pragma unroll
    for (int j = 0; j < 8; ++j) {
      const int row = q0 + j + 8 * hi;
      float e0 = sv[0][j] * scale; if (s0 + col > row)      e0 = -1e30f;
      float e1 = sv[1][j] * scale; if (s0 + 16 + col > row) e1 = -1e30f;
      float rm = fmaxf(e0, e1);
      rm = fmaxf(rm, __shfl_xor(rm, 1, 32));
      rm = fmaxf(rm, __shfl_xor(rm, 2, 32));
      rm = fmaxf(rm, __shfl_xor(rm, 4, 32));
      rm = fmaxf(rm, __shfl_xor(rm, 8, 32));
      float mnew = fmaxf(mrun[j], rm);
      float f  = __expf(mrun[j] - mnew);
      float p0 = __expf(e0 - mnew);
      float p1 = __expf(e1 - mnew);
      float ps = p0 + p1;
      ps += __shfl_xor(ps, 1, 32);
      ps += __shfl_xor(ps, 2, 32);
      ps += __shfl_xor(ps, 4, 32);
      ps += __shfl_xor(ps, 8, 32);
      lrun[j] = lrun[j] * f + ps;
      mrun[j] = mnew;
#pragma unroll
      for (int nt = 0; nt < 4; ++nt) oacc[nt][j] *= f;
      pLds[(j + 8 * hi) * 32 + col]      = (bf16)p0;
      pLds[(j + 8 * hi) * 32 + 16 + col] = (bf16)p1;
    }
    __syncthreads();
    // O += P(16x32) @ V(32x64)
    const bf16* pp = &pLds[col * 32 + ka];
    v16bf pa = cat16(*(const v8bf*)pp, *(const v8bf*)(pp + 16));
#pragma unroll
    for (int nt = 0; nt < 4; ++nt) {
      const bf16* vp = vt + ((size_t)bh * D_ + nt * 16 + col) * T_ + s0 + kb;
      v16bf vf = cat16(*(const v8bf*)vp, *(const v8bf*)(vp + 16));
      oacc[nt] = __builtin_amdgcn_wmma_f32_16x16x32_bf16(
          false, pa, false, vf, (short)0, oacc[nt], false, false);
    }
    __syncthreads();
  }
#pragma unroll
  for (int nt = 0; nt < 4; ++nt)
#pragma unroll
    for (int j = 0; j < 8; ++j)
      obh[((size_t)bh * T_ + q0 + j + 8 * hi) * D_ + nt * 16 + col] =
          oacc[nt][j] / lrun[j];
}

// ---------------- LN over gathered heads -> bf16 row-major ----------------
__global__ __launch_bounds__(256)
void ln_o(const float* __restrict__ obh, const float* __restrict__ g,
          const float* __restrict__ bb, bf16* __restrict__ oln) {
  const int bid = blockIdx.x;
  const int t = bid & (T_ - 1);
  const int b = bid >> 10;
  __shared__ float r1[256], r2[256];
  float s = 0.f, s2 = 0.f;
  for (int c = threadIdx.x; c < C_; c += 256) {
    float v = obh[((size_t)(b * H_ + (c >> 6)) * T_ + t) * D_ + (c & 63)];
    s += v; s2 += v * v;
  }
  r1[threadIdx.x] = s; r2[threadIdx.x] = s2;
  __syncthreads();
  for (int st = 128; st > 0; st >>= 1) {
    if (threadIdx.x < st) { r1[threadIdx.x] += r1[threadIdx.x + st];
                            r2[threadIdx.x] += r2[threadIdx.x + st]; }
    __syncthreads();
  }
  const float mu = r1[0] / C_;
  const float rinv = rsqrtf(r2[0] / C_ - mu * mu + 1e-5f);
  for (int c = threadIdx.x; c < C_; c += 256) {
    float v = obh[((size_t)(b * H_ + (c >> 6)) * T_ + t) * D_ + (c & 63)];
    oln[(size_t)bid * C_ + c] = (bf16)((v - mu) * rinv * g[c] + bb[c]);
  }
}

// ---------------- host launch ----------------
extern "C" void kernel_launch(void* const* d_in, const int* in_sizes, int n_in,
                              void* d_out, int out_size, void* d_ws, size_t ws_size,
                              hipStream_t stream) {
  const float* x     = (const float*)d_in[0];
  const float* shift = (const float*)d_in[1];
  const float* tmx   = (const float*)d_in[2];
  const float* tmr   = (const float*)d_in[3];
  const float* tmk   = (const float*)d_in[4];
  const float* tmv   = (const float*)d_in[5];
  const float* w1    = (const float*)d_in[6];
  const float* w2    = (const float*)d_in[7];
  const float* Wq    = (const float*)d_in[8];
  const float* Wk    = (const float*)d_in[9];
  const float* Wv    = (const float*)d_in[10];
  const float* Wo    = (const float*)d_in[11];
  const float* g_r   = (const float*)d_in[12];
  const float* b_r   = (const float*)d_in[13];
  const float* g_k   = (const float*)d_in[14];
  const float* b_k   = (const float*)d_in[15];
  const float* g_v   = (const float*)d_in[16];
  const float* b_v   = (const float*)d_in[17];
  const float* g_x   = (const float*)d_in[18];
  const float* b_x   = (const float*)d_in[19];
  const float* cosb  = (const float*)d_in[20];
  const float* sinb  = (const float*)d_in[21];
  float* out = (float*)d_out;

  const size_t MB = 1ull << 20;
  char* W = (char*)d_ws;
  float* dxprev = (float*)(W + 0 * MB);     // 16 MB
  float* xxx    = (float*)(W + 16 * MB);    // 16 MB
  float* hbuf   = (float*)(W + 32 * MB);    // 0.75 MB
  bf16*  xq     = (bf16*)(W + 33 * MB);     // 8 MB
  bf16*  xk     = (bf16*)(W + 41 * MB);     // 8 MB
  bf16*  xv     = (bf16*)(W + 49 * MB);     // 8 MB
  bf16*  WqT    = (bf16*)(W + 57 * MB);     // 8 MB
  bf16*  WkT    = (bf16*)(W + 65 * MB);     // 8 MB
  bf16*  WvT    = (bf16*)(W + 73 * MB);     // 8 MB
  bf16*  WoT    = (bf16*)(W + 81 * MB);     // 8 MB
  float* pre    = (float*)(W + 89 * MB);    // 16 MB (reused q/k/v)
  bf16*  qbh    = (bf16*)(W + 105 * MB);    // 8 MB
  bf16*  kbh    = (bf16*)(W + 113 * MB);    // 8 MB
  bf16*  vt     = (bf16*)(W + 121 * MB);    // 8 MB
  float* obh    = (float*)(W + 129 * MB);   // 16 MB
  bf16*  oln    = (bf16*)(W + 145 * MB);    // 8 MB  (total 153 MB)

  prep_shift<<<(M_ * C_) / 256, 256, 0, stream>>>(x, shift, tmx, dxprev, xxx);

  dim3 tb(32, 8), tg(C_ / 32, C_ / 32);
  transpose_to_bf16<<<tg, tb, 0, stream>>>(Wq, WqT, C_, C_);
  transpose_to_bf16<<<tg, tb, 0, stream>>>(Wk, WkT, C_, C_);
  transpose_to_bf16<<<tg, tb, 0, stream>>>(Wv, WvT, C_, C_);
  transpose_to_bf16<<<tg, tb, 0, stream>>>(Wo, WoT, C_, C_);

  lora1<<<M_, 128, 0, stream>>>(xxx, w1, hbuf);
  mix_qkv<<<M_, 256, 0, stream>>>(x, dxprev, hbuf, w2, tmr, tmk, tmv, xq, xk, xv);

  dim3 gg(C_ / 64, M_ / 64);
  gemm_bf16<<<gg, 32, 0, stream>>>(xq, WqT, pre, M_, C_, C_);
  ln_rope_qk<<<M_, 256, 0, stream>>>(pre, g_r, b_r, cosb, sinb, qbh);
  gemm_bf16<<<gg, 32, 0, stream>>>(xk, WkT, pre, M_, C_, C_);
  ln_rope_qk<<<M_, 256, 0, stream>>>(pre, g_k, b_k, cosb, sinb, kbh);
  gemm_bf16<<<gg, 32, 0, stream>>>(xv, WvT, pre, M_, C_, C_);
  ln_v<<<M_, 256, 0, stream>>>(pre, g_v, b_v, vt);

  dim3 ag(T_ / 16, H_, B_);
  attn_flash<<<ag, 32, 0, stream>>>(qbh, kbh, vt, obh);

  ln_o<<<M_, 256, 0, stream>>>(obh, g_x, b_x, oln);
  gemm_bf16<<<gg, 32, 0, stream>>>(oln, WoT, out, M_, C_, C_);
}
